// MultiAttentionResNet6EsDot_2_11304353923150
// MI455X (gfx1250) — matrix operand order repaired and checked by reading, not compiled, not run on hardware
//
#include <hip/hip_runtime.h>
#include <hip/hip_bf16.h>

typedef __attribute__((ext_vector_type(16))) _Float16 v16h;
typedef __attribute__((ext_vector_type(8)))  _Float16 v8h;
typedef __attribute__((ext_vector_type(8)))  float    v8f;

#define BN_ 2
#define UC 8
#define FC 64
#define AUX 5
#define HW 65536            // 256*256
#define NPIX 131072         // 2*65536

// ---------------------------------------------------------------------------
// Async global->LDS copy helpers (CDNA5 TDM-adjacent async path, ASYNCcnt)
// GVS mode: mem_addr = SADDR(u64) + VADDR(i32) ; VDST = per-lane LDS byte addr
// ---------------------------------------------------------------------------
__device__ __forceinline__ void async_g2l_b128(unsigned lds_byte, int g_byte_off,
                                               const void* base) {
  asm volatile("global_load_async_to_lds_b128 %0, %1, %2"
               :
               : "v"(lds_byte), "v"(g_byte_off), "s"(base)
               : "memory");
}
__device__ __forceinline__ void wait_async0() {
#if __has_builtin(__builtin_amdgcn_s_wait_asynccnt)
  __builtin_amdgcn_s_wait_asynccnt(0);
#else
  asm volatile("s_wait_asynccnt 0x0" ::: "memory");
#endif
}

// ---------------------------------------------------------------------------
// Front: uf = conv3(u, fu_w) [AUX ch], pf = conv3(pan, fp_w) [3 ch]  (f32 NCHW)
// ---------------------------------------------------------------------------
__global__ void k_front(const float* __restrict__ u, const float* __restrict__ pan,
                        const float* __restrict__ fu_w, const float* __restrict__ fp_w,
                        float* __restrict__ uf, float* __restrict__ pf) {
  int i = blockIdx.x * 256 + threadIdx.x;
  if (i >= NPIX) return;
  int b = i >> 16, rem = i & 65535, y = rem >> 8, x = rem & 255;
  for (int o = 0; o < AUX; ++o) {
    float s = 0.f;
    for (int ci = 0; ci < UC; ++ci)
      for (int ky = -1; ky <= 1; ++ky)
        for (int kx = -1; kx <= 1; ++kx) {
          int yy = y + ky, xx = x + kx;
          if (yy < 0 || yy > 255 || xx < 0 || xx > 255) continue;
          s += u[((b * UC + ci) << 16) + (yy << 8) + xx] *
               fu_w[((o * UC + ci) * 3 + (ky + 1)) * 3 + (kx + 1)];
        }
    uf[((b * AUX + o) << 16) + rem] = s;
  }
  for (int o = 0; o < 3; ++o) {
    float s = 0.f;
    for (int ky = -1; ky <= 1; ++ky)
      for (int kx = -1; kx <= 1; ++kx) {
        int yy = y + ky, xx = x + kx;
        if (yy < 0 || yy > 255 || xx < 0 || xx > 255) continue;
        s += pan[(b << 16) + (yy << 8) + xx] * fp_w[(o * 3 + (ky + 1)) * 3 + (kx + 1)];
      }
    pf[((b * 3 + o) << 16) + rem] = s;
  }
}

// ---------------------------------------------------------------------------
// phi = 1x1(phi_w, concat(in0,in1));  g = 1x1(g_w, uf)
// ---------------------------------------------------------------------------
__global__ void k_phig(const float* __restrict__ in0, int c0,
                       const float* __restrict__ in1, int c1,
                       const float* __restrict__ phi_w, const float* __restrict__ g_w,
                       const float* __restrict__ uf,
                       float* __restrict__ phi, float* __restrict__ g) {
  int i = blockIdx.x * 256 + threadIdx.x;
  if (i >= NPIX) return;
  int b = i >> 16, rem = i & 65535;
  float feat[8];
  int cph = c0 + c1;
  for (int c = 0; c < c0; ++c) feat[c] = in0[((b * c0 + c) << 16) + rem];
  for (int c = 0; c < c1; ++c) feat[c0 + c] = in1[((b * c1 + c) << 16) + rem];
  for (int o = 0; o < cph; ++o) {
    float s = 0.f;
    for (int c = 0; c < cph; ++c) s += phi_w[o * cph + c] * feat[c];
    phi[((b * cph + o) << 16) + rem] = s;
  }
  float fa[AUX];
  for (int c = 0; c < AUX; ++c) fa[c] = uf[((b * AUX + c) << 16) + rem];
  for (int o = 0; o < AUX; ++o) {
    float s = 0.f;
    for (int c = 0; c < AUX; ++c) s += g_w[o * AUX + c] * fa[c];
    g[((b * AUX + o) << 16) + rem] = s;
  }
}

__global__ void k_fill(float* __restrict__ p, const float* __restrict__ val, int n) {
  int i = blockIdx.x * 256 + threadIdx.x;
  if (i < n) p[i] = val[0];
}

// ---------------------------------------------------------------------------
// Local-window attention head (WIN=7), LDS-tiled; accumulates mlp_w[hk]*head.
// ---------------------------------------------------------------------------
template <int C, int P>
__global__ __launch_bounds__(256) void k_attn(const float* __restrict__ phi,
                                              const float* __restrict__ g,
                                              const float* __restrict__ mlp_w, int hk,
                                              float* __restrict__ u_att) {
  constexpr int PP = P / 2;
  constexpr int HALO = 4;
  constexpr int TS = 16 + 2 * HALO;  // 24
  constexpr int GS = 16 + 6;         // 22
  __shared__ float sphi[C][TS][TS];
  __shared__ float sg[AUX][GS][GS];
  int b = blockIdx.z;
  int ty0 = blockIdx.y * 16, tx0 = blockIdx.x * 16;
  int tid = threadIdx.x;
  for (int idx = tid; idx < C * TS * TS; idx += 256) {
    int c = idx / (TS * TS), r = idx % (TS * TS);
    int ry = r / TS, rx = r % TS;
    int yy = ty0 + ry - HALO, xx = tx0 + rx - HALO;
    float v = 0.f;
    if (yy >= 0 && yy < 256 && xx >= 0 && xx < 256)
      v = phi[((b * C + c) << 16) + (yy << 8) + xx];
    sphi[c][ry][rx] = v;
  }
  for (int idx = tid; idx < AUX * GS * GS; idx += 256) {
    int c = idx / (GS * GS), r = idx % (GS * GS);
    int ry = r / GS, rx = r % GS;
    int yy = ty0 + ry - 3, xx = tx0 + rx - 3;
    float v = 0.f;
    if (yy >= 0 && yy < 256 && xx >= 0 && xx < 256)
      v = g[((b * AUX + c) << 16) + (yy << 8) + xx];
    sg[c][ry][rx] = v;
  }
  __syncthreads();
  int ly = tid >> 4, lx = tid & 15;
  int y = ty0 + ly, x = tx0 + lx;
  float th[C * P * P];
#pragma unroll
  for (int c = 0; c < C; ++c)
#pragma unroll
    for (int py = 0; py < P; ++py)
#pragma unroll
      for (int px = 0; px < P; ++px)
        th[(c * P + py) * P + px] = sphi[c][ly + HALO + py - PP][lx + HALO + px - PP];
  float att[49];
  float mx = -1e30f;
  for (int o = 0; o < 49; ++o) {
    int dy = o / 7 - 3, dx = o % 7 - 3;
    int ky = y + dy, kx = x + dx;
    float dot = 0.f;
    if (ky >= 0 && ky < 256 && kx >= 0 && kx < 256) {
#pragma unroll
      for (int c = 0; c < C; ++c)
#pragma unroll
        for (int py = 0; py < P; ++py)
#pragma unroll
          for (int px = 0; px < P; ++px)
            dot += th[(c * P + py) * P + px] *
                   sphi[c][ly + HALO + dy + py - PP][lx + HALO + dx + px - PP];
    }
    att[o] = dot;
    mx = fmaxf(mx, dot);
  }
  float sum = 0.f;
  for (int o = 0; o < 49; ++o) { att[o] = __expf(att[o] - mx); sum += att[o]; }
  float inv = 1.f / sum;
  float hw = mlp_w[hk];
  for (int oc = 0; oc < AUX; ++oc) {
    float s = 0.f;
    for (int o = 0; o < 49; ++o) {
      int dy = o / 7 - 3, dx = o % 7 - 3;
      s += att[o] * sg[oc][ly + 3 + dy][lx + 3 + dx];
    }
    u_att[((b * AUX + oc) << 16) + (y << 8) + x] += hw * s * inv;
  }
}

// ---------------------------------------------------------------------------
// u (f32 NCHW, 8ch) -> uh (f16 NHWC, padded to 32 ch)
// ---------------------------------------------------------------------------
__global__ void k_u2h(const float* __restrict__ u, _Float16* __restrict__ uh) {
  int p = blockIdx.x * 256 + threadIdx.x;
  if (p >= NPIX) return;
  int b = p >> 16, rem = p & 65535;
  for (int c = 0; c < 32; ++c) {
    float v = (c < UC) ? u[((b * UC + c) << 16) + rem] : 0.f;
    uh[(size_t)p * 32 + c] = (_Float16)v;
  }
}

// ---------------------------------------------------------------------------
// Repack weights OIHW f32 -> f16 [tap][64][cinp] (zero-padded oc/ci)
// ---------------------------------------------------------------------------
__global__ void k_wrepack(const float* __restrict__ w, _Float16* __restrict__ wr,
                          int cout, int cin, int cinp) {
  int i = blockIdx.x * 256 + threadIdx.x;
  int total = 9 * 64 * cinp;
  if (i >= total) return;
  int pos = i / (64 * cinp);
  int rem = i - pos * 64 * cinp;
  int oc = rem / cinp;
  int ci = rem - oc * cinp;
  float v = 0.f;
  if (oc < cout && ci < cin) v = w[(oc * cin + ci) * 9 + pos];
  wr[i] = (_Float16)v;
}

// ---------------------------------------------------------------------------
// Implicit-GEMM 3x3 conv as 9 shifted 1x1 GEMMs via WMMA f32<-f16.
// Activations f16 NHWC (CINP ch/pixel); LDS staging uses async global->LDS
// b128 copies (ASYNCcnt) -- no VGPR round trip. Weights pre-repacked f16.
// Block = 8 waves = 128 pixels of one row x one 16-channel N tile.
// ---------------------------------------------------------------------------
template <int CIN>
__global__ __launch_bounds__(256) void k_conv3_wmma(
    const _Float16* __restrict__ in, const _Float16* __restrict__ wr,
    const float* __restrict__ bias, const _Float16* __restrict__ resid,
    _Float16* __restrict__ out, int cout, int cpout, int do_relu) {
  constexpr int NCH = (CIN + 31) / 32;
  constexpr int CINP = NCH * 32;
  __shared__ __attribute__((aligned(16))) _Float16 si[130 * CINP];
  __shared__ __attribute__((aligned(16))) _Float16 sw[9 * 16 * CINP];
  int tid = threadIdx.x;
  int n0 = blockIdx.y * 16;
  int m0blk = blockIdx.x * 128;
  int b = m0blk >> 16;
  int y0 = (m0blk & 65535) >> 8;
  int x0 = m0blk & 255;  // 0 or 128
  // async-copy weight slice: sw[pos][n][ci] <- wr[pos][n0+n][ci]
  unsigned swbase = (unsigned)(size_t)(void*)sw;  // low 32 bits = LDS offset
  constexpr int WV = 9 * 16 * CINP / 8;
  for (int v = tid; v < WV; v += 256) {
    int h0 = v * 8;
    int pos = h0 / (16 * CINP);
    int rem = h0 - pos * (16 * CINP);
    int gofs = (int)(((size_t)(pos * 64 + n0) * CINP + rem) * 2);
    async_g2l_b128(swbase + (unsigned)(h0 * 2), gofs, wr);
  }
  unsigned sibase = (unsigned)(size_t)(void*)si;
  int wave = tid >> 5, lane = tid & 31;
  int laneHi = lane >> 4, l15 = lane & 15;
  v8f acc = {};
#pragma unroll
  for (int dy = 0; dy < 3; ++dy) {
    int yy = y0 + dy - 1;
    bool rowok = (yy >= 0) && (yy < 256);
    __syncthreads();  // prior iteration's LDS reads complete
    if (rowok) {
      // one 130-col input row, contiguous in NHWC: async vector copy + halo fill
      int pixm1 = ((b << 16) + (yy << 8) + x0) - 1;
      constexpr int RV = 130 * CINP / 8;
      for (int v = tid; v < RV; v += 256) {
        int h0 = v * 8;
        int c = h0 / CINP;  // column (all 8 halfs in same column)
        int xx = x0 + c - 1;
        if (xx >= 0 && xx < 256) {
          async_g2l_b128(sibase + (unsigned)(h0 * 2),
                         (int)(((size_t)pixm1 * CINP + h0) * 2), in);
        } else {
          v8h z = {};
          *(v8h*)(si + h0) = z;
        }
      }
    }
    wait_async0();  // all our async copies (incl. weights) landed in LDS
    __syncthreads();
    if (rowok) {
#pragma unroll
      for (int dx = 0; dx < 3; ++dx) {
        const _Float16* ap = si + (size_t)(wave * 16 + l15 + dx) * CINP;
        const _Float16* bp = sw + (size_t)((dy * 3 + dx) * 16 + l15) * CINP;
#pragma unroll
        for (int kc = 0; kc < NCH; ++kc) {
          int kb = kc * 32;
          v8h alo = *(const v8h*)(ap + kb + laneHi * 8);
          v8h ahi = *(const v8h*)(ap + kb + 16 + laneHi * 8);
          v8h blo = *(const v8h*)(bp + kb + laneHi * 8);
          v8h bhi = *(const v8h*)(bp + kb + 16 + laneHi * 8);
          v16h a, bf;
#pragma unroll
          for (int e = 0; e < 8; ++e) {
            a[e] = alo[e];
            a[8 + e] = ahi[e];
            bf[e] = blo[e];
            bf[8 + e] = bhi[e];
          }
          acc = __builtin_amdgcn_wmma_f32_16x16x32_f16(false, a, false, bf, (short)0,
                                                       acc, false, false);
        }
      }
    }
  }
  // epilogue: lane<16: M=r ; lane>=16: M=8+r ; N = lane&15
  int n = n0 + l15;
  if (n < cout) {
    float bv = bias ? bias[n] : 0.f;
    int m0 = m0blk + wave * 16;
#pragma unroll
    for (int r = 0; r < 8; ++r) {
      int mm = m0 + r + (laneHi << 3);
      float v = acc[r] + bv;
      size_t oidx = (size_t)mm * cpout + n;
      if (resid) v += (float)resid[oidx];
      if (do_relu) v = fmaxf(v, 0.f);
      out[oidx] = (_Float16)v;
    }
  }
}

// ---------------------------------------------------------------------------
// u_att (f32 NCHW 5ch) -> dst f16 NHWC at channel offset
// ---------------------------------------------------------------------------
__global__ void k_att2x(const float* __restrict__ u_att, _Float16* __restrict__ dst,
                        int cp, int coff) {
  int i = blockIdx.x * 256 + threadIdx.x;
  if (i >= NPIX * AUX) return;
  int p = i / AUX, c = i - p * AUX;
  int b = p >> 16, rem = p & 65535;
  dst[(size_t)p * cp + coff + c] = (_Float16)u_att[((b * AUX + c) << 16) + rem];
}

// R(96) = concat(Y f16 NHWC 64, u_att f32 NCHW 5, zeros 27)
__global__ void k_concat_R(const _Float16* __restrict__ Y, const float* __restrict__ u_att,
                           _Float16* __restrict__ R) {
  int p = blockIdx.x * 256 + threadIdx.x;
  if (p >= NPIX) return;
  int b = p >> 16, rem = p & 65535;
  for (int c = 0; c < 64; ++c) R[(size_t)p * 96 + c] = Y[(size_t)p * 64 + c];
  for (int c = 0; c < AUX; ++c)
    R[(size_t)p * 96 + 64 + c] = (_Float16)u_att[((b * AUX + c) << 16) + rem];
  for (int c = 64 + AUX; c < 96; ++c) R[(size_t)p * 96 + c] = (_Float16)0.f;
}

// ---------------------------------------------------------------------------
// BatchNorm training-mode stats on f16 NHWC: one block per channel.
// ---------------------------------------------------------------------------
__global__ void k_bn_stats_h(const _Float16* __restrict__ x, int cp,
                             float* __restrict__ stats) {
  int ch = blockIdx.x;
  __shared__ float ssum[256], ssq[256];
  float s = 0.f, q = 0.f;
  for (int p = threadIdx.x; p < NPIX; p += 256) {
    float v = (float)x[(size_t)p * cp + ch];
    s += v;
    q += v * v;
  }
  ssum[threadIdx.x] = s;
  ssq[threadIdx.x] = q;
  __syncthreads();
  for (int off = 128; off > 0; off >>= 1) {
    if (threadIdx.x < off) {
      ssum[threadIdx.x] += ssum[threadIdx.x + off];
      ssq[threadIdx.x] += ssq[threadIdx.x + off];
    }
    __syncthreads();
  }
  if (threadIdx.x == 0) {
    float mean = ssum[0] * (1.f / NPIX);
    float var = ssq[0] * (1.f / NPIX) - mean * mean;
    stats[2 * ch] = mean;
    stats[2 * ch + 1] = rsqrtf(var + 1e-5f);
  }
}

// y = relu(bn(x)), f16 NHWC -> f16 NHWC (cp must be power of two)
__global__ void k_bn_apply_h(const _Float16* __restrict__ x, const float* __restrict__ stats,
                             const float* __restrict__ gamma, const float* __restrict__ beta,
                             _Float16* __restrict__ y, int cp) {
  int i = blockIdx.x * 256 + threadIdx.x;
  if (i >= NPIX * cp) return;
  int ch = i & (cp - 1);
  float v = ((float)x[i] - stats[2 * ch]) * stats[2 * ch + 1] * gamma[ch] + beta[ch];
  y[i] = (_Float16)fmaxf(v, 0.f);
}

// out(f32 NCHW) = relu(bn(S f16 NHWC 8)) + u
__global__ void k_bn_final(const _Float16* __restrict__ S, const float* __restrict__ stats,
                           const float* __restrict__ gamma, const float* __restrict__ beta,
                           const float* __restrict__ u, float* __restrict__ out) {
  int p = blockIdx.x * 256 + threadIdx.x;
  if (p >= NPIX) return;
  int b = p >> 16, rem = p & 65535;
  for (int c = 0; c < UC; ++c) {
    float v = ((float)S[(size_t)p * 8 + c] - stats[2 * c]) * stats[2 * c + 1] * gamma[c] +
              beta[c];
    v = fmaxf(v, 0.f);
    int idx = ((b * UC + c) << 16) + rem;
    out[idx] = v + u[idx];
  }
}

// ---------------------------------------------------------------------------
extern "C" void kernel_launch(void* const* d_in, const int* in_sizes, int n_in,
                              void* d_out, int out_size, void* d_ws, size_t ws_size,
                              hipStream_t stream) {
  const float* u      = (const float*)d_in[0];
  const float* pan    = (const float*)d_in[1];
  const float* h1_phi = (const float*)d_in[2];
  const float* h1_g   = (const float*)d_in[3];
  const float* h2_phi = (const float*)d_in[4];
  const float* h2_g   = (const float*)d_in[5];
  const float* h3_phi = (const float*)d_in[6];
  const float* h3_g   = (const float*)d_in[7];
  const float* mlp_w  = (const float*)d_in[8];
  const float* mlp_b  = (const float*)d_in[9];
  const float* rn_w   = (const float*)d_in[10];
  const float* r1c1_w = (const float*)d_in[11];
  const float* r1c1_b = (const float*)d_in[12];
  const float* r1c2_w = (const float*)d_in[13];
  const float* r1c2_b = (const float*)d_in[14];
  const float* r2c1_w = (const float*)d_in[15];
  const float* r2c1_b = (const float*)d_in[16];
  const float* r2c2_w = (const float*)d_in[17];
  const float* r2c2_b = (const float*)d_in[18];
  const float* r3c1_w = (const float*)d_in[19];
  const float* r3c1_b = (const float*)d_in[20];
  const float* r3c2_w = (const float*)d_in[21];
  const float* r3c2_b = (const float*)d_in[22];
  const float* fu_w   = (const float*)d_in[23];
  const float* fp_w   = (const float*)d_in[24];
  const float* rc1_w  = (const float*)d_in[25];
  const float* bn1_g  = (const float*)d_in[26];
  const float* bn1_b  = (const float*)d_in[27];
  const float* rc2_w  = (const float*)d_in[28];
  const float* bn2_g  = (const float*)d_in[29];
  const float* bn2_b  = (const float*)d_in[30];
  float* out = (float*)d_out;

  // ---- workspace layout: f32 region, then 64B-aligned f16 region ----
  char* wsb = (char*)d_ws;
  size_t off = 0;
  auto allocf = [&](size_t nfloats) {
    float* p = (float*)(wsb + off);
    off += nfloats * 4;
    return p;
  };
  float* uf    = allocf((size_t)AUX * NPIX);
  float* pf    = allocf((size_t)3 * NPIX);
  float* phi   = allocf((size_t)8 * NPIX);
  float* g     = allocf((size_t)AUX * NPIX);
  float* u_att = allocf((size_t)AUX * NPIX);
  float* stats = allocf(256);
  off = (off + 63) & ~(size_t)63;
  auto alloch = [&](size_t nh) {
    _Float16* p = (_Float16*)(wsb + off);
    off += ((nh * 2 + 63) & ~(size_t)63);
    return p;
  };
  _Float16* uh = alloch((size_t)NPIX * 32);
  _Float16* X  = alloch((size_t)NPIX * 64);
  _Float16* T  = alloch((size_t)NPIX * 64);
  _Float16* Y  = alloch((size_t)NPIX * 64);
  _Float16* R  = alloch((size_t)NPIX * 96);
  _Float16* S  = alloch((size_t)NPIX * 8);
  _Float16* wr = alloch((size_t)9 * 64 * 96);

  const int EW = (NPIX + 255) / 256;
  dim3 blk(256);

  // front features + heads (f32 path)
  k_front<<<EW, blk, 0, stream>>>(u, pan, fu_w, fp_w, uf, pf);
  k_fill<<<(AUX * NPIX + 255) / 256, blk, 0, stream>>>(u_att, mlp_b, AUX * NPIX);
  dim3 agrid(16, 16, 2);
  k_phig<<<EW, blk, 0, stream>>>(pf, 3, nullptr, 0, h1_phi, h1_g, uf, phi, g);
  k_attn<3, 3><<<agrid, blk, 0, stream>>>(phi, g, mlp_w, 0, u_att);
  k_phig<<<EW, blk, 0, stream>>>(uf, AUX, nullptr, 0, h2_phi, h2_g, uf, phi, g);
  k_attn<5, 1><<<agrid, blk, 0, stream>>>(phi, g, mlp_w, 1, u_att);
  k_phig<<<EW, blk, 0, stream>>>(uf, AUX, pf, 3, h3_phi, h3_g, uf, phi, g);
  k_attn<8, 3><<<agrid, blk, 0, stream>>>(phi, g, mlp_w, 2, u_att);

  // f16 NHWC input copy of u
  k_u2h<<<EW, blk, 0, stream>>>(u, uh);

  auto rp = [&](const float* w, int cout, int cin, int cinp) {
    int total = 9 * 64 * cinp;
    k_wrepack<<<(total + 255) / 256, blk, 0, stream>>>(w, wr, cout, cin, cinp);
  };

  dim3 cgrid4(NPIX / 128, 4);
  dim3 cgrid1(NPIX / 128, 1);

  // X[0:59] = conv3(u); X[59:64] = u_att
  rp(rn_w, FC - AUX, UC, 32);
  k_conv3_wmma<UC><<<cgrid4, blk, 0, stream>>>(uh, wr, nullptr, nullptr, X, FC - AUX, FC, 0);
  k_att2x<<<(AUX * NPIX + 255) / 256, blk, 0, stream>>>(u_att, X, FC, FC - AUX);

  // three residual blocks (64->64, WMMA)
  rp(r1c1_w, FC, FC, FC);
  k_conv3_wmma<FC><<<cgrid4, blk, 0, stream>>>(X, wr, r1c1_b, nullptr, T, FC, FC, 1);
  rp(r1c2_w, FC, FC, FC);
  k_conv3_wmma<FC><<<cgrid4, blk, 0, stream>>>(T, wr, r1c2_b, X, Y, FC, FC, 1);
  rp(r2c1_w, FC, FC, FC);
  k_conv3_wmma<FC><<<cgrid4, blk, 0, stream>>>(Y, wr, r2c1_b, nullptr, T, FC, FC, 1);
  rp(r2c2_w, FC, FC, FC);
  k_conv3_wmma<FC><<<cgrid4, blk, 0, stream>>>(T, wr, r2c2_b, Y, X, FC, FC, 1);
  rp(r3c1_w, FC, FC, FC);
  k_conv3_wmma<FC><<<cgrid4, blk, 0, stream>>>(X, wr, r3c1_b, nullptr, T, FC, FC, 1);
  rp(r3c2_w, FC, FC, FC);
  k_conv3_wmma<FC><<<cgrid4, blk, 0, stream>>>(T, wr, r3c2_b, X, Y, FC, FC, 1);

  // R = concat(Y, u_att, 0-pad) -> conv 69->64 -> BN1 -> relu
  k_concat_R<<<EW, blk, 0, stream>>>(Y, u_att, R);
  rp(rc1_w, FC, FC + AUX, 96);
  k_conv3_wmma<FC + AUX><<<cgrid4, blk, 0, stream>>>(R, wr, nullptr, nullptr, T, FC, FC, 0);
  k_bn_stats_h<<<FC, blk, 0, stream>>>(T, FC, stats);
  k_bn_apply_h<<<(FC * NPIX + 255) / 256, blk, 0, stream>>>(T, stats, bn1_g, bn1_b, X, FC);

  // conv 64->8 -> BN2 -> relu -> + u  (final f32 NCHW output)
  rp(rc2_w, UC, FC, FC);
  k_conv3_wmma<FC><<<cgrid1, blk, 0, stream>>>(X, wr, nullptr, nullptr, S, UC, UC, 0);
  k_bn_stats_h<<<UC, blk, 0, stream>>>(S, UC, stats + 128);
  k_bn_final<<<EW, blk, 0, stream>>>(S, stats + 128, bn2_g, bn2_b, u, out);
}
